// MoleculeGenerator_9028021256836
// MI455X (gfx1250) — matrix operand order repaired and checked by reading, not compiled, not run on hardware
//
#include <hip/hip_runtime.h>
#include <hip/hip_bf16.h>

typedef __bf16 bf16;
typedef __attribute__((ext_vector_type(16))) __bf16 v16bf;
typedef __attribute__((ext_vector_type(8)))  __bf16 v8bf;
typedef __attribute__((ext_vector_type(8)))  float  v8f;

#define H 256
#define BN 4096
#define NF 131072
#define NR 262144
#define V 8001
#define VPAD 8064          // multiple of 128 (block tile N of the 2x4-wave GEMM)
#define KIN 768
#define DIST2 36.0f

__device__ __forceinline__ int lower_bound_i(const int* __restrict__ a, int n, int key) {
    int lo = 0, hi = n;
    while (lo < hi) { int mid = (lo + hi) >> 1; if (a[mid] < key) lo = mid + 1; else hi = mid; }
    return lo;
}

// Segment-sum atoms -> pred_vecs[:, 0:256] (bf16) and center positions.
__global__ void k_atom_segsum(const float* __restrict__ h_ctx,
                              const float* __restrict__ pos_ctx,
                              const int*   __restrict__ batch,
                              bf16*  __restrict__ pred_vecs,
                              float* __restrict__ center) {
    const int b = blockIdx.x;
    const int t = threadIdx.x;
    const int lo = lower_bound_i(batch, NF, b);
    const int hi = lower_bound_i(batch, NF, b + 1);
    float acc = 0.0f;
    for (int r = lo; r < hi; ++r) {
        if (r + 4 < hi) __builtin_prefetch(&h_ctx[(size_t)(r + 4) * H + t], 0, 0);
        acc += h_ctx[(size_t)r * H + t];
    }
    pred_vecs[(size_t)b * KIN + t] = (bf16)acc;
    if (t < 3) {
        float p = 0.0f;
        for (int r = lo; r < hi; ++r) p += pos_ctx[(size_t)r * 3 + t];
        float cnt = (float)(hi - lo);
        center[b * 4 + t] = p / fmaxf(cnt, 1.0f);
    }
}

// emb_weight[current_wid] -> pred_vecs[:, 256:512]
__global__ void k_motif_gather(const float* __restrict__ emb,
                               const int*   __restrict__ wid,
                               bf16* __restrict__ pred_vecs) {
    const int idx = blockIdx.x * blockDim.x + threadIdx.x; // B*H
    const int b = idx >> 8, t = idx & 255;
    pred_vecs[(size_t)b * KIN + H + t] = (bf16)emb[(size_t)wid[b] * H + t];
}

// Masked residue segment-sum -> pred_vecs[:, 512:768]
__global__ void k_residue_segsum(const float* __restrict__ h_res,
                                 const float* __restrict__ res_pos,
                                 const int*   __restrict__ batch,
                                 const float* __restrict__ center,
                                 bf16* __restrict__ pred_vecs) {
    const int b = blockIdx.x;
    const int t = threadIdx.x;
    const int lo = lower_bound_i(batch, NR, b);
    const int hi = lower_bound_i(batch, NR, b + 1);
    const float cx = center[b * 4 + 0];
    const float cy = center[b * 4 + 1];
    const float cz = center[b * 4 + 2];
    float acc = 0.0f;
    for (int r = lo; r < hi; ++r) {
        if (r + 4 < hi) __builtin_prefetch(&h_res[(size_t)(r + 4) * H + t], 0, 0);
        const float dx = res_pos[(size_t)r * 9 + 3] - cx;
        const float dy = res_pos[(size_t)r * 9 + 4] - cy;
        const float dz = res_pos[(size_t)r * 9 + 5] - cz;
        if (dx * dx + dy * dy + dz * dz < DIST2)
            acc += h_res[(size_t)r * H + t];
    }
    pred_vecs[(size_t)b * KIN + 2 * H + t] = (bf16)acc;
}

// dst[n*K+k] = (bf16)src[k*N+n]   (K x N f32 -> N x K bf16)
__global__ void k_transpose_bf16(const float* __restrict__ src, bf16* __restrict__ dst,
                                 int K, int N) {
    const int idx = blockIdx.x * blockDim.x + threadIdx.x;
    if (idx >= K * N) return;
    const int k = idx / N, n = idx % N;
    dst[(size_t)n * K + k] = (bf16)src[idx];
}

// emb f32 [V,H] -> bf16 [VPAD,H], zero padded rows
__global__ void k_convert_emb(const float* __restrict__ src, bf16* __restrict__ dst) {
    const int idx = blockIdx.x * blockDim.x + threadIdx.x;
    if (idx >= VPAD * H) return;
    const int row = idx >> 8;
    dst[idx] = (row < V) ? (bf16)src[idx] : (bf16)0.0f;
}

// C[M,N] = A[M,K] * B[N,K]^T (+bias, relu).
// Block = 256 threads = 8 waves arranged 2(M) x 4(N); block tile 128x128.
// Wave tile 64(M) x 32(N) = 4x2 register blocking: per K=32 step,
// 8 WMMAs fed by 12 b128 loads (4 A frags + 2 B frags).
// Requires: M % 128 == 0, Npad % 128 == 0.
// A lane layout: row = lane&15; a[0..7]  = A[row][kb + (lane>>4)*8 .. +7],
//                              a[8..15] = A[row][kb+16 + (lane>>4)*8 .. +7]
// B lane layout: n = lane&15;  b[0..7]  = B[n][kb + (lane>>4)*8 .. +7],
//                              b[8..15] = B[n][kb+16 + (lane>>4)*8 .. +7]
__global__ void __launch_bounds__(256)
k_gemm_nt_bf16(const bf16* __restrict__ A, const bf16* __restrict__ Bm,
               const float* __restrict__ bias,
               float* __restrict__ outF, bf16* __restrict__ outB,
               int M, int Npad, int Nvalid, int K, int ldc, int relu) {
    const int lane = threadIdx.x & 31;
    const int wv   = threadIdx.x >> 5;          // 0..7
    const int nbn  = Npad >> 7;                 // 128-wide N block tiles
    const int bm   = blockIdx.x / nbn;
    const int bn   = blockIdx.x % nbn;
    const int m0   = (bm << 7) + ((wv >> 2) << 6);   // 2 M-wave-groups of 64
    const int n0   = (bn << 7) + ((wv & 3) << 5);    // 4 N-wave-groups of 32
    const int row  = lane & 15;
    const int half = lane >> 4;

    const bf16* Arow[4];
#pragma unroll
    for (int i = 0; i < 4; ++i) Arow[i] = A + (size_t)(m0 + i * 16 + row) * K;
    const bf16* Brow[2];
#pragma unroll
    for (int j = 0; j < 2; ++j) Brow[j] = Bm + (size_t)(n0 + j * 16 + row) * K;

    v8f acc[4][2];
#pragma unroll
    for (int i = 0; i < 4; ++i)
#pragma unroll
        for (int j = 0; j < 2; ++j) acc[i][j] = (v8f){};

    for (int kb = 0; kb < K; kb += 32) {
        v16bf a[4], b[2];
#pragma unroll
        for (int i = 0; i < 4; ++i) {
            v8bf a_lo = *(const v8bf*)(Arow[i] + kb + half * 8);
            v8bf a_hi = *(const v8bf*)(Arow[i] + kb + 16 + half * 8);
            a[i] = __builtin_shufflevector(a_lo, a_hi,
                       0,1,2,3,4,5,6,7,8,9,10,11,12,13,14,15);
        }
#pragma unroll
        for (int j = 0; j < 2; ++j) {
            v8bf b_lo = *(const v8bf*)(Brow[j] + kb + half * 8);
            v8bf b_hi = *(const v8bf*)(Brow[j] + kb + 16 + half * 8);
            b[j] = __builtin_shufflevector(b_lo, b_hi,
                       0,1,2,3,4,5,6,7,8,9,10,11,12,13,14,15);
        }
#pragma unroll
        for (int i = 0; i < 4; ++i)
#pragma unroll
            for (int j = 0; j < 2; ++j)
                acc[i][j] = __builtin_amdgcn_wmma_f32_16x16x32_bf16(
                                false, a[i], false, b[j], (short)0, acc[i][j],
                                false, false);
    }

#pragma unroll
    for (int j = 0; j < 2; ++j) {
        const int n = n0 + j * 16 + row;
        if (n < Nvalid) {
            const float bv = bias ? bias[n] : 0.0f;
#pragma unroll
            for (int i = 0; i < 4; ++i) {
#pragma unroll
                for (int v = 0; v < 8; ++v) {
                    const int m = m0 + i * 16 + v + half * 8;
                    float r = acc[i][j][v] + bv;
                    if (relu) r = fmaxf(r, 0.0f);
                    if (outF) outF[(size_t)m * ldc + n] = r;
                    if (outB) outB[(size_t)m * ldc + n] = (bf16)r;
                }
            }
        }
    }
}

// Per-row top-5 + deterministic pick -> preds
__global__ void k_topk(const float* __restrict__ scores, float* __restrict__ preds) {
    __shared__ float sv[256 * 5];
    __shared__ int   si[256 * 5];
    const int b = blockIdx.x, t = threadIdx.x;
    float tv[5]; int ti[5];
#pragma unroll
    for (int j = 0; j < 5; ++j) { tv[j] = -3.4e38f; ti[j] = 0x7fffffff; }
    const float* rowp = scores + (size_t)b * V;
    for (int j = t; j < V; j += 256) {
        const float s = rowp[j];
        if (s > tv[4]) {
            int p = 4;
            while (p > 0 && s > tv[p - 1]) { tv[p] = tv[p - 1]; ti[p] = ti[p - 1]; --p; }
            tv[p] = s; ti[p] = j;
        }
    }
#pragma unroll
    for (int j = 0; j < 5; ++j) { sv[t * 5 + j] = tv[j]; si[t * 5 + j] = ti[j]; }
    for (int s = 128; s > 0; s >>= 1) {
        __syncthreads();
        if (t < s) {
            float cv[10]; int ci[10];
#pragma unroll
            for (int j = 0; j < 5; ++j) {
                cv[j] = sv[t * 5 + j];           ci[j] = si[t * 5 + j];
                cv[5 + j] = sv[(t + s) * 5 + j]; ci[5 + j] = si[(t + s) * 5 + j];
            }
#pragma unroll
            for (int j = 0; j < 5; ++j) {
                int best = j;
                for (int k = j + 1; k < 10; ++k)
                    if (cv[k] > cv[best] || (cv[k] == cv[best] && ci[k] < ci[best])) best = k;
                float fv = cv[j]; cv[j] = cv[best]; cv[best] = fv;
                int iv = ci[j]; ci[j] = ci[best]; ci[best] = iv;
                sv[t * 5 + j] = cv[j]; si[t * 5 + j] = ci[j];
            }
        }
    }
    __syncthreads();
    if (t == 0) {
        const unsigned u = ((unsigned)b * 2654435761u) >> 13;
        preds[b] = (float)si[u % 5u];
    }
}

extern "C" void kernel_launch(void* const* d_in, const int* in_sizes, int n_in,
                              void* d_out, int out_size, void* d_ws, size_t ws_size,
                              hipStream_t stream) {
    const float* h_ctx_focal  = (const float*)d_in[0];
    const float* pos_ctx      = (const float*)d_in[1];
    const int*   current_wid  = (const int*)d_in[2];
    const int*   atoms_batch  = (const int*)d_in[3];
    const float* h_residue    = (const float*)d_in[4];
    const float* residue_pos  = (const float*)d_in[5];
    const int*   aa_batch     = (const int*)d_in[6];
    const float* emb_weight   = (const float*)d_in[7];
    const float* mlp_w1       = (const float*)d_in[8];
    const float* mlp_b1       = (const float*)d_in[9];
    const float* mlp_w2       = (const float*)d_in[10];
    const float* mlp_b2       = (const float*)d_in[11];

    float* scores = (float*)d_out;                   // [B, V]
    float* preds  = (float*)d_out + (size_t)BN * V;  // [B]

    char* w = (char*)d_ws;
    bf16*  pred_vecs = (bf16*)w;                 w += (size_t)BN * KIN * 2;   // 6 MB
    bf16*  w1t       = (bf16*)w;                 w += (size_t)H * KIN * 2;    // 384 KB
    bf16*  w2t       = (bf16*)w;                 w += (size_t)H * H * 2;      // 128 KB
    bf16*  embp      = (bf16*)w;                 w += (size_t)VPAD * H * 2;   // ~4.1 MB
    bf16*  h_bf      = (bf16*)w;                 w += (size_t)BN * H * 2;     // 2 MB
    bf16*  mh_bf     = (bf16*)w;                 w += (size_t)BN * H * 2;     // 2 MB
    float* center    = (float*)w;                                            // 64 KB

    // Stage 1: feature assembly (bandwidth-bound, no atomics)
    k_atom_segsum<<<BN, 256, 0, stream>>>(h_ctx_focal, pos_ctx, atoms_batch,
                                          pred_vecs, center);
    k_motif_gather<<<(BN * H) / 256, 256, 0, stream>>>(emb_weight, current_wid, pred_vecs);
    k_residue_segsum<<<BN, 256, 0, stream>>>(h_residue, residue_pos, aa_batch,
                                             center, pred_vecs);

    // Stage 2: weight conversion to bf16 N x K layout
    k_transpose_bf16<<<(KIN * H + 255) / 256, 256, 0, stream>>>(mlp_w1, w1t, KIN, H);
    k_transpose_bf16<<<(H * H + 255) / 256, 256, 0, stream>>>(mlp_w2, w2t, H, H);
    k_convert_emb<<<(VPAD * H) / 256, 256, 0, stream>>>(emb_weight, embp);

    // Stage 3: WMMA GEMM chain (128x128 block tile, 64x32 per wave, 4x2 frags)
    {   // h = relu(pred_vecs @ w1 + b1) -> bf16
        const int grid = (BN / 128) * (H / 128);
        k_gemm_nt_bf16<<<grid, 256, 0, stream>>>(
            pred_vecs, w1t, mlp_b1, nullptr, h_bf, BN, H, H, KIN, H, 1);
    }
    {   // mh = h @ w2 + b2 -> bf16
        const int grid = (BN / 128) * (H / 128);
        k_gemm_nt_bf16<<<grid, 256, 0, stream>>>(
            h_bf, w2t, mlp_b2, nullptr, mh_bf, BN, H, H, H, H, 0);
    }
    {   // pred_scores = mh @ emb^T -> f32 out (guarded to 8001 cols)
        const int grid = (BN / 128) * (VPAD / 128);
        k_gemm_nt_bf16<<<grid, 256, 0, stream>>>(
            mh_bf, embp, nullptr, scores, nullptr, BN, VPAD, V, H, V, 0);
    }

    // Stage 4: top-k + deterministic selection
    k_topk<<<BN, 256, 0, stream>>>(scores, preds);
}
